// ViT_53584011985165
// MI455X (gfx1250) — matrix operand order repaired and checked by reading, not compiled, not run on hardware
//
#include <hip/hip_runtime.h>
#include <hip/hip_bf16.h>
#include <math.h>

// ---------------- model constants ----------------
#define D_MODEL 512
#define N_HEADS 8
#define HEAD_DIM 64
#define NBATCH 4
#define SEQ 4096
#define NLAYERS 12
#define NLROWS (NBATCH * SEQ)                 // 16384 rows
static const long NLC = (long)NBATCH * SEQ * D_MODEL;  // 8388608 elements

#define USE_TDM 1

// ---------------- types ----------------
typedef __bf16 bf16_t;
typedef __attribute__((ext_vector_type(16))) __bf16 v16bf;
typedef __attribute__((ext_vector_type(8)))  __bf16 v8bf;
typedef __attribute__((ext_vector_type(4)))  __bf16 v4bf;
typedef __attribute__((ext_vector_type(8)))  float  v8f;
typedef __attribute__((ext_vector_type(4)))  float  f32x4;
typedef __attribute__((ext_vector_type(4)))  unsigned int u32x4;
typedef __attribute__((ext_vector_type(8)))  int i32x8;
typedef __attribute__((ext_vector_type(4)))  int i32x4;

enum { EPI_NONE = 0, EPI_ELU1 = 1, EPI_RELU = 2 };

// =====================================================================
// TDM: issue a 2D tile DMA (rows x kElems bf16) global -> LDS.
// D# built per CDNA5 ISA 8.3/8.4: group0 = {flags, lds_addr, global_addr,
// type=2}; group1 = {mask/data_size, dims, tile dims, dim0 stride}.
// Tensor dims are set equal to the tile (tile fully in-bounds by
// construction) so no OOB handling is required.
// This toolchain's builtin is the 6-arg form:
//   (uint32x4 g0, int32x8 g1, int32x4 g2, int32x4 g3, int32x8 pad, int cpol)
// =====================================================================
__device__ __forceinline__ void tdm_load_tile(unsigned lds_off, const bf16_t* gptr,
                                              int kElems, int rows, long strideElems)
{
    unsigned long long ga = (unsigned long long)(size_t)gptr;
    u32x4 g0;
    g0[0] = 1u;                                     // count=1, user desc, no gather
    g0[1] = lds_off;                                // lds_addr (bytes)
    g0[2] = (unsigned)(ga & 0xffffffffu);           // global_addr[31:0]
    g0[3] = (unsigned)((ga >> 32) & 0x01ffffffu) | 0x80000000u; // addr[56:32] | type=2

    unsigned ukd = (unsigned)kElems;
    unsigned urw = (unsigned)rows;
    unsigned long long st = (unsigned long long)strideElems;
    i32x8 g1;
    g1[0] = (int)(1u << 16);                        // workgroup_mask=0, data_size=1 (2B)
    g1[1] = (int)((ukd & 0xffffu) << 16);           // tensor_dim0[15:0] @ bits63:48
    g1[2] = (int)(((ukd >> 16) & 0xffffu) | ((urw & 0xffffu) << 16)); // dim0[31:16], dim1[15:0]
    g1[3] = (int)(((urw >> 16) & 0xffffu) | ((ukd & 0xffffu) << 16)); // dim1[31:16], tile_dim0
    g1[4] = (int)(urw & 0xffffu);                   // tile_dim1 (tile_dim2=0)
    g1[5] = (int)(st & 0xffffffffu);                // tensor_dim0_stride[31:0]
    g1[6] = (int)((st >> 32) & 0xffffu);            // stride[47:32]; dim1_stride[15:0]=0
    g1[7] = 0;
    i32x4 z4 = {0, 0, 0, 0};
    i32x8 z8 = {0, 0, 0, 0, 0, 0, 0, 0};
#if USE_TDM
    __builtin_amdgcn_tensor_load_to_lds(g0, g1, z4, z4, z8, 0);
#else
    (void)g0; (void)g1; (void)z4; (void)z8;
#endif
}

// =====================================================================
// Fast bf16 GEMM: C = epi(alpha * A@B), A bf16 row-major [M x K] (lda),
// optional A2 concat along K at K1, B given pre-transposed bf16 BT[N x K]
// (ldb). Requires M%128==0, N%128==0, K%32==0, K1%32==0.
// Block 256 thr (8 waves 4x2), tile 128x128x32, double-buffered LDS,
// tile staging via Tensor Data Mover (waves 0/1), 8 WMMA / wave / K-step.
// =====================================================================
#define FBM 128
#define FBN 128
#define FBK 32

__global__ void __launch_bounds__(256)
fast_gemm_kernel(const bf16_t* __restrict__ A, int lda,
                 const bf16_t* __restrict__ A2, int lda2, int K1,
                 const bf16_t* __restrict__ BT, int ldb,
                 float* __restrict__ Cf, bf16_t* __restrict__ Cb, int ldc,
                 int K, float alpha, int epi)
{
    __shared__ bf16_t As[2][FBM * FBK];
    __shared__ bf16_t Bs[2][FBN * FBK];

    const int tid  = threadIdx.x;
    const int lane = tid & 31;
    const int wid  = tid >> 5;
    const int wm   = wid >> 1;            // 0..3  (32 rows each)
    const int wn   = wid & 1;             // 0..1  (64 cols each)
    const long m0  = (long)blockIdx.y * FBM;
    const long n0  = (long)blockIdx.x * FBN;

    const int mrow  = lane & 15;
    const int khalf = lane >> 4;

    v8f acc[2][4] = {};

    auto stage = [&](int buf, int k0) {
#if USE_TDM
        if (tid < 32) {
            const bf16_t* src; long st;
            if (A2 != nullptr && k0 >= K1) { src = A2 + m0 * (long)lda2 + (k0 - K1); st = lda2; }
            else                           { src = A  + m0 * (long)lda  + k0;        st = lda;  }
            tdm_load_tile((unsigned)(size_t)&As[buf][0], src, FBK, FBM, st);
        } else if (tid < 64) {
            tdm_load_tile((unsigned)(size_t)&Bs[buf][0], BT + n0 * (long)ldb + k0, FBK, FBN, ldb);
        }
#else
        {   // manual staging: 32B per thread per tile, fully vectorized
            const int row = tid >> 1, seg = tid & 1;
            const bf16_t* src;
            if (A2 != nullptr && k0 >= K1) { src = A2 + (m0 + row) * (long)lda2 + (k0 - K1); }
            else                           { src = A  + (m0 + row) * (long)lda  + k0;        }
            *(v8bf*)&As[buf][row * FBK + seg * 16]     = *(const v8bf*)(src + seg * 16);
            *(v8bf*)&As[buf][row * FBK + seg * 16 + 8] = *(const v8bf*)(src + seg * 16 + 8);
            const bf16_t* sb = BT + (n0 + row) * (long)ldb + k0;
            *(v8bf*)&Bs[buf][row * FBK + seg * 16]     = *(const v8bf*)(sb + seg * 16);
            *(v8bf*)&Bs[buf][row * FBK + seg * 16 + 8] = *(const v8bf*)(sb + seg * 16 + 8);
        }
#endif
    };

    stage(0, 0);
    const int nk = K / FBK;

    for (int i = 0; i < nk; ++i) {
        const int cur = i & 1;
#if USE_TDM
        __builtin_amdgcn_s_wait_tensorcnt(0);   // waves 0/1: my DMA done; others: no-op
#endif
        __syncthreads();                        // tile visible to all waves
        if (i + 1 < nk) stage(cur ^ 1, (i + 1) * FBK);

        const bf16_t* Ab = &As[cur][0];
        const bf16_t* Bb = &Bs[cur][0];
#pragma unroll
        for (int tm = 0; tm < 2; ++tm) {
            const bf16_t* ar = Ab + (wm * 32 + tm * 16 + mrow) * FBK + khalf * 16;
            v8bf alo = *(const v8bf*)ar;
            v8bf ahi = *(const v8bf*)(ar + 8);
            v16bf af = __builtin_shufflevector(alo, ahi,
                        0,1,2,3,4,5,6,7,8,9,10,11,12,13,14,15);
#pragma unroll
            for (int tn = 0; tn < 4; ++tn) {
                const bf16_t* br = Bb + (wn * 64 + tn * 16 + mrow) * FBK + khalf * 16;
                v16bf bv = *(const v16bf*)br;
                acc[tm][tn] = __builtin_amdgcn_wmma_f32_16x16x32_bf16(
                    false, af, false, bv, (short)0, acc[tm][tn], false, false);
            }
        }
    }

    // epilogue (C/D layout: vgpr r -> M = khalf*8 + r, lane%16 -> N)
#pragma unroll
    for (int tm = 0; tm < 2; ++tm) {
#pragma unroll
        for (int tn = 0; tn < 4; ++tn) {
            const long gm0 = m0 + wm * 32 + tm * 16 + khalf * 8;
            const long gn  = n0 + wn * 64 + tn * 16 + mrow;
#pragma unroll
            for (int r = 0; r < 8; ++r) {
                float x = acc[tm][tn][r] * alpha;
                if (epi == EPI_ELU1)      x = (x > 0.f) ? (x + 1.f) : __expf(x);
                else if (epi == EPI_RELU) x = fmaxf(x, 0.f);
                if (Cb) Cb[(gm0 + r) * ldc + gn] = (bf16_t)x;
                else    Cf[(gm0 + r) * ldc + gn] = x;
            }
        }
    }
}

// =====================================================================
// Generic guarded f32-input GEMM (kept for the small batched head GEMMs:
// KV = K^T@V (M=N=64,K=4096) and msg = Q@KV (M=4096,N=K=64)).
// =====================================================================
#define BM 128
#define BN 64
#define BK 32

__global__ void __launch_bounds__(256)
gemm_wmma_kernel(const float* __restrict__ A, long sAm, long sAk,
                 const float* __restrict__ B, long sBk, long sBn,
                 float* __restrict__ C, long sCm, long sCn,
                 int M, int N, int K,
                 int HH, long bsA1, long bsA2, long bsB1, long bsB2,
                 long bsC1, long bsC2)
{
    __shared__ bf16_t As[BM][BK];
    __shared__ bf16_t BsT[BN][BK];

    const int tid  = threadIdx.x;
    const int lane = tid & 31;
    const int wid  = tid >> 5;
    const int wm   = wid >> 1;
    const int wn   = wid & 1;

    const int bz = blockIdx.z;
    A += (long)(bz / HH) * bsA1 + (long)(bz % HH) * bsA2;
    B += (long)(bz / HH) * bsB1 + (long)(bz % HH) * bsB2;
    C += (long)(bz / HH) * bsC1 + (long)(bz % HH) * bsC2;

    const long m0 = (long)blockIdx.y * BM;
    const long n0 = (long)blockIdx.x * BN;

    v8f acc[2][2] = {};
    const int mrow  = lane & 15;
    const int khalf = lane >> 4;

    for (int k0 = 0; k0 < K; k0 += BK) {
        {
            const int kk = tid & 31;
            const int ib = tid >> 5;
            const int kg = k0 + kk;
#pragma unroll 4
            for (int j = 0; j < 16; ++j) {
                const int  i  = ib + 8 * j;
                const long gm = m0 + i;
                float v = 0.f;
                if (gm < M) v = A[gm * sAm + (long)kg * sAk];
                As[i][kk] = (bf16_t)v;
            }
            if (k0 + BK < K && m0 + ib < M)
                __builtin_prefetch(&A[(m0 + ib) * sAm + (long)(k0 + BK + kk) * sAk], 0, 1);
        }
        {
            const int j  = tid & 63;
            const int kb = tid >> 6;
            const long gn = n0 + j;
#pragma unroll 4
            for (int c = 0; c < 8; ++c) {
                const int kk = kb + 4 * c;
                float v = 0.f;
                if (gn < N) v = B[(long)(k0 + kk) * sBk + gn * sBn];
                BsT[j][kk] = (bf16_t)v;
            }
        }
        __syncthreads();
#pragma unroll
        for (int tm = 0; tm < 2; ++tm) {
            v16bf af;
            const int row = wm * 32 + tm * 16 + mrow;
            const int kb  = khalf * 16;
#pragma unroll
            for (int e = 0; e < 8; ++e) { af[e] = As[row][kb + e]; af[8 + e] = As[row][kb + 8 + e]; }
#pragma unroll
            for (int tn = 0; tn < 2; ++tn) {
                v16bf bv;
                const int col = wn * 32 + tn * 16 + mrow;
                const int kb2 = khalf * 16;
#pragma unroll
                for (int e = 0; e < 16; ++e) bv[e] = BsT[col][kb2 + e];
                acc[tm][tn] = __builtin_amdgcn_wmma_f32_16x16x32_bf16(
                    false, af, false, bv, (short)0, acc[tm][tn], false, false);
            }
        }
        __syncthreads();
    }
#pragma unroll
    for (int tm = 0; tm < 2; ++tm)
#pragma unroll
        for (int tn = 0; tn < 2; ++tn) {
            const long baseM = m0 + wm * 32 + tm * 16 + khalf * 8;
            const long gn    = n0 + wn * 32 + tn * 16 + mrow;
            if (gn < N) {
#pragma unroll
                for (int r = 0; r < 8; ++r) {
                    const long gm = baseM + r;
                    if (gm < M) C[gm * sCm + gn * sCn] = acc[tm][tn][r];
                }
            }
        }
}

// ---------------- elementwise / reduction kernels ----------------

// flat f32 -> bf16 (vectorized x4)
__global__ void cvt_bf16_kernel(const f32x4* __restrict__ s, v4bf* __restrict__ d)
{
    const long i = (long)blockIdx.x * blockDim.x + threadIdx.x;
    f32x4 x = s[i];
    v4bf y;
#pragma unroll
    for (int e = 0; e < 4; ++e) y[e] = (bf16_t)x[e];
    d[i] = y;
}

// WT[n*K + k] = bf16(W[k*N + n])
__global__ void cvt_tr_kernel(const float* __restrict__ W, bf16_t* __restrict__ WT,
                              int K, int N)
{
    const long i = (long)blockIdx.x * blockDim.x + threadIdx.x;
    const int n = (int)(i / K), k = (int)(i % K);
    WT[i] = (bf16_t)W[(long)k * N + n];
}

// Ksum[n,h,d] = sum_l K[n,l,h,d]
__global__ void ksum_kernel(const float* __restrict__ Kf, float* __restrict__ Ksum)
{
    const int b = blockIdx.x;            // n*H + h
    const int d = threadIdx.x;           // 0..63
    const int n = b / N_HEADS, h = b % N_HEADS;
    const float* base = Kf + (long)n * SEQ * D_MODEL + h * HEAD_DIM + d;
    float s = 0.f;
    for (int l = 0; l < SEQ; ++l) s += base[(long)l * D_MODEL];
    Ksum[b * HEAD_DIM + d] = s;
}

// out_bf16[n,l,h,:] = msg[n,l,h,:] * S / (Q . Ksum + eps)
__global__ void zscale_kernel(const float* __restrict__ Q,
                              const float* __restrict__ Ksum,
                              const float* __restrict__ MsgIn,
                              bf16_t* __restrict__ MsgOut)
{
    const long idx = (long)blockIdx.x * blockDim.x + threadIdx.x;   // NB*SEQ*H
    const long nl  = idx / N_HEADS;
    const int  h   = (int)(idx % N_HEADS);
    const long n   = nl / SEQ;
    const float* q  = Q    + nl * D_MODEL + h * HEAD_DIM;
    const float* ks = Ksum + (n * N_HEADS + h) * HEAD_DIM;
    float dot = 0.f;
#pragma unroll 8
    for (int d = 0; d < HEAD_DIM; ++d) dot += q[d] * ks[d];
    const float z = (float)SEQ / (dot + 1e-6f);
    const float* mi = MsgIn  + nl * D_MODEL + h * HEAD_DIM;
    bf16_t*      mo = MsgOut + nl * D_MODEL + h * HEAD_DIM;
#pragma unroll 8
    for (int d = 0; d < HEAD_DIM; ++d) mo[d] = (bf16_t)(mi[d] * z);
}

// LayerNorm over C=512; writes f32 (optionally + residual) and/or bf16
__global__ void __launch_bounds__(256)
ln_kernel(const float* __restrict__ src, const float* __restrict__ g,
          const float* __restrict__ b, const float* __restrict__ resid,
          float* __restrict__ dstF, bf16_t* __restrict__ dstB)
{
    __shared__ float red[256];
    const long row = blockIdx.x;
    const float* s = src + row * D_MODEL;
    const int t = threadIdx.x;
    const float x0 = s[t], x1 = s[t + 256];

    red[t] = x0 + x1;
    __syncthreads();
    for (int off = 128; off > 0; off >>= 1) { if (t < off) red[t] += red[t + off]; __syncthreads(); }
    const float mu = red[0] * (1.f / D_MODEL);
    __syncthreads();

    const float d0 = x0 - mu, d1 = x1 - mu;
    red[t] = d0 * d0 + d1 * d1;
    __syncthreads();
    for (int off = 128; off > 0; off >>= 1) { if (t < off) red[t] += red[t + off]; __syncthreads(); }
    const float rstd = rsqrtf(red[0] * (1.f / D_MODEL) + 1e-5f);

    float y0 = d0 * rstd * g[t]       + b[t];
    float y1 = d1 * rstd * g[t + 256] + b[t + 256];
    if (resid) { y0 += resid[row * D_MODEL + t]; y1 += resid[row * D_MODEL + t + 256]; }
    if (dstF) { dstF[row * D_MODEL + t] = y0; dstF[row * D_MODEL + t + 256] = y1; }
    if (dstB) { dstB[row * D_MODEL + t] = (bf16_t)y0; dstB[row * D_MODEL + t + 256] = (bf16_t)y1; }
}

// ---------------- host side ----------------
struct LayerP {
    const float *Wq, *Wk, *Wv, *Wm, *W1, *W2, *g1, *b1, *g2, *b2;
};

struct Bufs {
    float *bufQ, *bufK, *bufV, *bufMf, *bufT, *bufKV, *Ksum;
    bf16_t *Xbf, *Sbf, *Mbf, *LNbf, *Hbf;
    bf16_t *WqT, *WkT, *WvT, *WmT, *W1T, *W2T;
};

static inline void fast_gemm(hipStream_t st, const bf16_t* A, int lda,
                             const bf16_t* A2, int lda2, int K1,
                             const bf16_t* BT, int ldb,
                             float* Cf, bf16_t* Cb, int ldc,
                             int M, int N, int K, float alpha, int epi)
{
    dim3 grid(N / FBN, M / FBM);
    fast_gemm_kernel<<<grid, dim3(256), 0, st>>>(A, lda, A2, lda2, K1, BT, ldb,
                                                 Cf, Cb, ldc, K, alpha, epi);
}

static inline void cvt(hipStream_t st, const float* s, bf16_t* d, long n)
{
    cvt_bf16_kernel<<<(int)(n / 4 / 256), 256, 0, st>>>((const f32x4*)s, (v4bf*)d);
}

static inline void cvt_tr(hipStream_t st, const float* W, bf16_t* WT, int K, int N)
{
    cvt_tr_kernel<<<(int)(((long)K * N) / 256), 256, 0, st>>>(W, WT, K, N);
}

static void encoder_layer(hipStream_t st, float* X, const float* SRC, const LayerP& p,
                          const Bufs& w)
{
    const int C = D_MODEL, L = SEQ, Nn = NBATCH, H = N_HEADS, D = HEAD_DIM;
    const int NL = Nn * L;

    // inputs -> bf16
    cvt(st, X, w.Xbf, NLC);
    const bf16_t* Sbf = w.Xbf;
    if (SRC != X) { cvt(st, SRC, w.Sbf, NLC); Sbf = w.Sbf; }
    // weights -> bf16 transposed
    cvt_tr(st, p.Wq, w.WqT, C, C);
    cvt_tr(st, p.Wk, w.WkT, C, C);
    cvt_tr(st, p.Wv, w.WvT, C, C);
    cvt_tr(st, p.Wm, w.WmT, C, C);
    cvt_tr(st, p.W1, w.W1T, 2 * C, 2 * C);
    cvt_tr(st, p.W2, w.W2T, 2 * C, C);

    // Q = elu(X@Wq)+1 ; K = elu(SRC@Wk)+1 ; V = (SRC@Wv)/S   (f32 outputs)
    fast_gemm(st, w.Xbf, C, nullptr, 0, 0, w.WqT, C, w.bufQ, nullptr, C,
              NL, C, C, 1.f, EPI_ELU1);
    fast_gemm(st, Sbf,   C, nullptr, 0, 0, w.WkT, C, w.bufK, nullptr, C,
              NL, C, C, 1.f, EPI_ELU1);
    fast_gemm(st, Sbf,   C, nullptr, 0, 0, w.WvT, C, w.bufV, nullptr, C,
              NL, C, C, 1.f / (float)L, EPI_NONE);

    ksum_kernel<<<Nn * H, D, 0, st>>>(w.bufK, w.Ksum);

    // KV[n,h] = K^T @ V (batched 32x: M=64,N=64,K=4096)
    {
        dim3 grid((D + BN - 1) / BN, (D + BM - 1) / BM, Nn * H);
        gemm_wmma_kernel<<<grid, dim3(256), 0, st>>>(
            w.bufK, 1, C, w.bufV, C, 1, w.bufKV, D, 1,
            D, D, L, H, (long)L * C, D, (long)L * C, D,
            (long)H * D * D, (long)D * D);
    }
    // msgPre = Q @ KV (batched 32x: M=4096,N=64,K=64)
    {
        dim3 grid((D + BN - 1) / BN, (L + BM - 1) / BM, Nn * H);
        gemm_wmma_kernel<<<grid, dim3(256), 0, st>>>(
            w.bufQ, C, 1, w.bufKV, D, 1, w.bufMf, C, 1,
            L, D, D, H, (long)L * C, D, (long)H * D * D, (long)D * D,
            (long)L * C, D);
    }

    // msg *= S / (Q.Ksum + eps)  -> bf16
    zscale_kernel<<<(NL * H) / 256, 256, 0, st>>>(w.bufQ, w.Ksum, w.bufMf, w.Mbf);

    // msg @ Wm -> f32 ; LN1 -> bf16
    fast_gemm(st, w.Mbf, C, nullptr, 0, 0, w.WmT, C, w.bufT, nullptr, C,
              NL, C, C, 1.f, EPI_NONE);
    ln_kernel<<<NL, 256, 0, st>>>(w.bufT, p.g1, p.b1, nullptr, nullptr, w.LNbf);

    // hid = relu([X | msgLN] @ W1) -> bf16 directly
    fast_gemm(st, w.Xbf, C, w.LNbf, C, C, w.W1T, 2 * C, nullptr, w.Hbf, 2 * C,
              NL, 2 * C, 2 * C, 1.f, EPI_RELU);
    // ffn2 -> f32
    fast_gemm(st, w.Hbf, 2 * C, nullptr, 0, 0, w.W2T, 2 * C, w.bufT, nullptr, C,
              NL, C, 2 * C, 1.f, EPI_NONE);

    // X = X + LN2(ffn)
    ln_kernel<<<NL, 256, 0, st>>>(w.bufT, p.g2, p.b2, X, X, nullptr);
}

extern "C" void kernel_launch(void* const* d_in, const int* in_sizes, int n_in,
                              void* d_out, int out_size, void* d_ws, size_t ws_size,
                              hipStream_t stream)
{
    const float* feat0 = (const float*)d_in[0];
    const float* feat1 = (const float*)d_in[1];
    const float* Wq = (const float*)d_in[2];
    const float* Wk = (const float*)d_in[3];
    const float* Wv = (const float*)d_in[4];
    const float* Wm = (const float*)d_in[5];
    const float* W1 = (const float*)d_in[6];
    const float* W2 = (const float*)d_in[7];
    const float* g1 = (const float*)d_in[8];
    const float* b1 = (const float*)d_in[9];
    const float* g2 = (const float*)d_in[10];
    const float* b2 = (const float*)d_in[11];

    const long C = D_MODEL;

    float* f0 = (float*)d_out;
    float* f1 = f0 + NLC;
    (void)hipMemcpyAsync(f0, feat0, NLC * sizeof(float), hipMemcpyDeviceToDevice, stream);
    (void)hipMemcpyAsync(f1, feat1, NLC * sizeof(float), hipMemcpyDeviceToDevice, stream);

    // ---- workspace carve: f32 region then bf16 region ----
    Bufs w;
    float* wsf = (float*)d_ws;
    w.bufQ  = wsf;               wsf += NLC;
    w.bufK  = wsf;               wsf += NLC;
    w.bufV  = wsf;               wsf += NLC;
    w.bufMf = wsf;               wsf += NLC;
    w.bufKV = wsf;               wsf += 32 * 64 * 64;
    w.Ksum  = wsf;               wsf += 32 * 64;
    w.bufT  = w.bufK;            // bufK dead before bufT is produced

    bf16_t* wsb = (bf16_t*)wsf;
    w.Xbf  = wsb;                wsb += NLC;
    w.Sbf  = wsb;                wsb += NLC;
    w.Mbf  = wsb;                wsb += NLC;
    w.LNbf = wsb;                wsb += NLC;
    w.Hbf  = wsb;                wsb += 2 * NLC;
    w.WqT  = wsb;                wsb += C * C;
    w.WkT  = wsb;                wsb += C * C;
    w.WvT  = wsb;                wsb += C * C;
    w.WmT  = wsb;                wsb += C * C;
    w.W1T  = wsb;                wsb += 4 * C * C;
    w.W2T  = wsb;                wsb += 2 * C * C;

    for (int i = 0; i < NLAYERS; ++i) {
        LayerP p;
        p.Wq = Wq + (long)i * C * C;
        p.Wk = Wk + (long)i * C * C;
        p.Wv = Wv + (long)i * C * C;
        p.Wm = Wm + (long)i * C * C;
        p.W1 = W1 + (long)i * 4 * C * C;
        p.W2 = W2 + (long)i * 2 * C * C;
        p.g1 = g1 + (long)i * C;  p.b1 = b1 + (long)i * C;
        p.g2 = g2 + (long)i * C;  p.b2 = b2 + (long)i * C;

        if (i % 2 == 0) {   // self
            encoder_layer(stream, f0, f0, p, w);
            encoder_layer(stream, f1, f1, p, w);
        } else {            // cross (feat1 sees updated feat0)
            encoder_layer(stream, f0, f1, p, w);
            encoder_layer(stream, f1, f0, p, w);
        }
    }
}
